// RITS_71983651881272
// MI455X (gfx1250) — matrix operand order repaired and checked by reading, not compiled, not run on hardware
//
#include <hip/hip_runtime.h>
#include <hip/hip_bf16.h>

typedef _Float16 half8 __attribute__((ext_vector_type(8)));
typedef _Float16 v16h  __attribute__((ext_vector_type(16)));
typedef float    v8f   __attribute__((ext_vector_type(8)));

#define B_    512
#define T_    1024
#define D_    64
#define H_    128
#define INSZ  130     // 2*D + 2
#define KTOT  288     // [x_c 64 | m 64 | d 1 | w 1 | pad 30][h 128]
#define NKC   9       // K chunks of 32
#define ASTR  296     // f16 row stride for A-stage and Wcat (bank-conflict pad)
#define XSTR  72      // f16 row stride for raw-x stage
#define XHSTR 68      // f32 row stride for x_h / x_f stages
#define BTILE 16
#define NWG   (B_/BTILE)   // 32 workgroups

// ---- LDS layout (bytes) ----
#define OFF_WCAT  0
#define SZ_WCAT   (512*ASTR*2)            // 303104
#define OFF_AST   (OFF_WCAT + SZ_WCAT)
#define SZ_AST    (BTILE*ASTR*2)          // 9472
#define OFF_XST   (OFF_AST + SZ_AST)
#define SZ_XST    (BTILE*XSTR*2)          // 2304
#define OFF_XH    (OFF_XST + SZ_XST)
#define SZ_XH     (BTILE*XHSTR*4)         // 4352
#define OFF_XF    (OFF_XH + SZ_XH)
#define SZ_XF     (BTILE*XHSTR*4)         // 4352
#define OFF_DROW  (OFF_XF + SZ_XF)
#define SZ_DROW   (2*16*4)                // double-buffered d per row
#define OFF_WROW  (OFF_DROW + SZ_DROW)
#define SZ_WROW   (2*16*4)                // double-buffered w_surv per row
#define OFF_LOSS  (OFF_WROW + SZ_WROW)
#define SZ_LOSS   16
#define SMEM_BYTES (OFF_LOSS + SZ_LOSS)   // 323856 <= 327680 (320 KB)

__device__ __forceinline__ v16h frag16(const _Float16* p0, const _Float16* p1) {
  half8 lo = *(const half8*)p0;
  half8 hi = *(const half8*)p1;
  return __builtin_shufflevector(lo, hi, 0,1,2,3,4,5,6,7,8,9,10,11,12,13,14,15);
}
__device__ __forceinline__ float sigf(float x) { return 1.0f / (1.0f + __expf(-x)); }

#define WMMA16(A, Bm, C) \
  __builtin_amdgcn_wmma_f32_16x16x32_f16(false, (A), false, (Bm), (short)0, (C), false, false)

__global__ void __launch_bounds__(512, 1)
rits_main(const float* __restrict__ data,   const float* __restrict__ masks,
          const float* __restrict__ deltas, const float* __restrict__ ages,
          const int*   __restrict__ clusters, const float* __restrict__ surv,
          const float* __restrict__ W_ih,   const float* __restrict__ W_hh,
          const float* __restrict__ b_ih,   const float* __restrict__ b_hh,
          const float* __restrict__ W_hist, const float* __restrict__ b_hist,
          const float* __restrict__ W_feat, const float* __restrict__ b_feat,
          const float* __restrict__ W_decay,const float* __restrict__ b_decay,
          float* __restrict__ out, float* __restrict__ ws)
{
  extern __shared__ char smem[];
  _Float16* Wcat = (_Float16*)(smem + OFF_WCAT);
  _Float16* Ast  = (_Float16*)(smem + OFF_AST);
  _Float16* Xst  = (_Float16*)(smem + OFF_XST);
  float* xhb   = (float*)(smem + OFF_XH);
  float* xfb   = (float*)(smem + OFF_XF);
  float* drow  = (float*)(smem + OFF_DROW);
  float* wrow  = (float*)(smem + OFF_WROW);
  float* lossb = (float*)(smem + OFF_LOSS);

  const int tid    = threadIdx.x;            // 0..511
  const int wave   = tid >> 5;               // 16 waves
  const int lane   = tid & 31;
  const int lane16 = lane & 15;
  const int half   = lane >> 4;
  const int b0     = blockIdx.x * BTILE;
  const int ol     = tid - 256;              // owner id for waves 8..15

  // ---------------- one-time init ----------------
  {
    // Wcat row n = tid : [W_ih[n][0..129] | zeros to 160 | W_hh[n][0..127] | pad]
    _Float16* wr = Wcat + tid * ASTR;
    const float* wih = W_ih + tid * INSZ;
    const float* whh = W_hh + tid * H_;
    for (int k = 0; k < INSZ; ++k)  wr[k] = (_Float16)wih[k];
    for (int k = INSZ; k < 160; ++k) wr[k] = (_Float16)0.f;
    for (int k = 0; k < H_; ++k)    wr[160 + k] = (_Float16)whh[k];
    for (int k = KTOT; k < ASTR; ++k) wr[k] = (_Float16)0.f;
  }
  if (tid < BTILE) {
    for (int k = INSZ; k < 160; ++k) Ast[tid*ASTR + k] = (_Float16)0.f;   // K pad inside inputs
    for (int k = KTOT; k < ASTR; ++k) Ast[tid*ASTR + k] = (_Float16)0.f;
    for (int k = D_; k < XSTR; ++k)  Xst[tid*XSTR + k] = (_Float16)0.f;
  }
  if (tid == 0) { lossb[0] = 0.f; lossb[1] = 0.f; }

  // Per-wave persistent register state
  v8f  h = {0,0,0,0,0,0,0,0};
  v8f  c = {0,0,0,0,0,0,0,0};
  float bias0 = 0.f, bias1 = 0.f, bias2 = 0.f, bias3 = 0.f;
  float Wd = 0.f, bd = 0.f, bcol = 0.f;
  v16h Bh0 = {}, Bh1 = {}, Bh2 = {}, Bh3 = {};   // W_hist B-fragments (waves 8..11)
  v16h Bf0 = {}, Bf1 = {};                        // W_feat B-fragments (waves 12..15)

  if (wave < 8) {
    const int jc = wave*16 + lane16;             // hidden column block owned by this wave
    Wd = W_decay[jc];  bd = b_decay[jc];
    bias0 = b_ih[jc        ] + b_hh[jc        ];
    bias1 = b_ih[jc + 128  ] + b_hh[jc + 128  ];
    bias2 = b_ih[jc + 256  ] + b_hh[jc + 256  ];
    bias3 = b_ih[jc + 384  ] + b_hh[jc + 384  ];
  } else if (wave < 12) {
    const int n = (wave - 8)*16 + lane16;        // output column of x_h
    #pragma unroll
    for (int kc = 0; kc < 4; ++kc) {
      const int k0 = kc*32 + half*8;
      v16h f = {};
      #pragma unroll
      for (int e = 0; e < 16; ++e) f[e] = (_Float16)W_hist[n*H_ + k0 + e + (e & 8)];
      if (kc == 0) Bh0 = f; else if (kc == 1) Bh1 = f; else if (kc == 2) Bh2 = f; else Bh3 = f;
    }
    bcol = b_hist[n];
  } else {
    const int n = (wave - 12)*16 + lane16;       // output column of x_f
    #pragma unroll
    for (int kc = 0; kc < 2; ++kc) {
      const int k0 = kc*32 + half*8;
      v16h f = {};
      #pragma unroll
      for (int e = 0; e < 16; ++e) f[e] = (_Float16)W_feat[n*D_ + k0 + e + (e & 8)];
      if (kc == 0) Bf0 = f; else Bf1 = f;
    }
    bcol = b_feat[n];
  }

  if (wave == 15 && lane < 16) {                 // preload d/w for t=0 (buffer 0)
    const int b = b0 + lane;
    drow[lane] = deltas[(size_t)b*T_ + 0];
    const float ag = ages[(size_t)b*T_ + 0];
    const int   cl = clusters[(size_t)b*T_ + 0];
    wrow[lane] = __expf(-surv[cl] * ag);
  }
  __syncthreads();

  // ---------------- recurrence over T ----------------
  for (int t = 0; t < T_; ++t) {
    const int buf = (t & 1) * 16;
    float xv[4], mv[4];
    v8f ga0 = {}, ga1 = {}, ga2 = {}, ga3 = {};

    // phase 0: gate waves apply decay to h and stage h' (f16) into A-stage h-region;
    //          owner waves issue this step's x/m global loads (consumed in phase 2).
    if (wave < 8) {
      const int jc = wave*16 + lane16;
      #pragma unroll
      for (int r = 0; r < 8; ++r) {
        const int M = r + 8*half;
        float a = drow[buf + M]*Wd + bd;
        a = a > 0.f ? a : 0.f;
        h[r] = h[r] * __expf(-a);
        Ast[M*ASTR + 160 + jc] = (_Float16)h[r];
      }
    } else {
      #pragma unroll
      for (int q = 0; q < 4; ++q) {
        const int e = q*256 + ol, r = e >> 6, col = e & 63;
        const size_t gi = ((size_t)(b0 + r)*T_ + t)*D_ + col;
        xv[q] = data[gi];  mv[q] = masks[gi];
      }
    }
    __syncthreads();

    // phase 1: x_h = h' @ W_hist^T + b_hist  (waves 8..11; all A-frags preloaded,
    //          4 back-to-back WMMAs chained on the accumulator)
    if (wave >= 8 && wave < 12) {
      const int n = (wave - 8)*16 + lane16;
      const _Float16* arow = &Ast[lane16*ASTR + 160 + half*8];
      v16h a0 = frag16(arow +  0, arow + 16);
      v16h a1 = frag16(arow + 32, arow + 48);
      v16h a2 = frag16(arow + 64, arow + 80);
      v16h a3 = frag16(arow + 96, arow + 112);
      v8f acc = {bcol,bcol,bcol,bcol,bcol,bcol,bcol,bcol};
      acc = WMMA16(a0, Bh0, acc);
      acc = WMMA16(a1, Bh1, acc);
      acc = WMMA16(a2, Bh2, acc);
      acc = WMMA16(a3, Bh3, acc);
      #pragma unroll
      for (int r = 0; r < 8; ++r) xhb[(r + 8*half)*XHSTR + n] = acc[r];
    } else if (wave < 8) {
      if (t + 1 < T_) {
        const float* src = (wave < 4) ? data : masks;
        const int cb = ((wave & 3) << 4) + (half << 3);
        __builtin_prefetch(&src[((size_t)(b0 + lane16)*T_ + (t + 1))*D_ + cb], 0, 1);
      }
    }
    __syncthreads();

    // phase 2: owners build A-stage: x_c, m, raw x (for x_f), d/w scalars
    if (wave >= 8) {
      #pragma unroll
      for (int q = 0; q < 4; ++q) {
        const int e = q*256 + ol, r = e >> 6, col = e & 63;
        const float m = mv[q];
        const float xc = m*xv[q] + (1.f - m)*xhb[r*XHSTR + col];
        Ast[r*ASTR + col]      = (_Float16)xc;
        Ast[r*ASTR + 64 + col] = (_Float16)m;
        Xst[r*XSTR + col]      = (_Float16)xv[q];
      }
      if (ol < 16) {
        Ast[ol*ASTR + 128] = (_Float16)drow[buf + ol];
        Ast[ol*ASTR + 129] = (_Float16)wrow[buf + ol];
      }
    }
    __syncthreads();

    // phase 3: gates = [inputs|h'] @ Wcat^T + bias  (waves 0..7)
    //          Software-pipelined: chunk kc+1 fragment loads are issued before the
    //          chunk-kc WMMAs so DS latency hides behind the matrix pipe.
    //          x_f = x @ W_feat^T + b_feat          (waves 12..15)
    if (wave < 8) {
      ga0 = (v8f){bias0,bias0,bias0,bias0,bias0,bias0,bias0,bias0};
      ga1 = (v8f){bias1,bias1,bias1,bias1,bias1,bias1,bias1,bias1};
      ga2 = (v8f){bias2,bias2,bias2,bias2,bias2,bias2,bias2,bias2};
      ga3 = (v8f){bias3,bias3,bias3,bias3,bias3,bias3,bias3,bias3};
      const int nb = wave*16 + lane16;
      const _Float16* arow = &Ast[lane16*ASTR + half*16];
      const _Float16* w0 = &Wcat[(size_t)(nb      )*ASTR + half*16];
      const _Float16* w1 = &Wcat[(size_t)(nb + 128)*ASTR + half*16];
      const _Float16* w2 = &Wcat[(size_t)(nb + 256)*ASTR + half*16];
      const _Float16* w3 = &Wcat[(size_t)(nb + 384)*ASTR + half*16];
      v16h aC  = frag16(arow, arow + 8);
      v16h b0C = frag16(w0, w0 + 8);
      v16h b1C = frag16(w1, w1 + 8);
      v16h b2C = frag16(w2, w2 + 8);
      v16h b3C = frag16(w3, w3 + 8);
      #pragma unroll
      for (int kc = 0; kc < NKC; ++kc) {
        v16h aN = aC, b0N = b0C, b1N = b1C, b2N = b2C, b3N = b3C;
        if (kc + 1 < NKC) {
          const int c0 = (kc + 1)*32;
          aN  = frag16(arow + c0, arow + c0 + 8);
          b0N = frag16(w0 + c0, w0 + c0 + 8);
          b1N = frag16(w1 + c0, w1 + c0 + 8);
          b2N = frag16(w2 + c0, w2 + c0 + 8);
          b3N = frag16(w3 + c0, w3 + c0 + 8);
        }
        ga0 = WMMA16(aC, b0C, ga0);
        ga1 = WMMA16(aC, b1C, ga1);
        ga2 = WMMA16(aC, b2C, ga2);
        ga3 = WMMA16(aC, b3C, ga3);
        aC = aN; b0C = b0N; b1C = b1N; b2C = b2N; b3C = b3N;
      }
    } else if (wave >= 12) {
      const int n = (wave - 12)*16 + lane16;
      const _Float16* xrow = &Xst[lane16*XSTR + half*8];
      v16h a0 = frag16(xrow +  0, xrow + 16);
      v16h a1 = frag16(xrow + 32, xrow + 48);
      v8f acc = {bcol,bcol,bcol,bcol,bcol,bcol,bcol,bcol};
      acc = WMMA16(a0, Bf0, acc);
      acc = WMMA16(a1, Bf1, acc);
      #pragma unroll
      for (int r = 0; r < 8; ++r) xfb[(r + 8*half)*XHSTR + n] = acc[r];
    }
    __syncthreads();

    // phase 4: LSTM cell update entirely in registers; stage new h (f16) for x_h2
    if (wave < 8) {
      const int jc = wave*16 + lane16;
      #pragma unroll
      for (int r = 0; r < 8; ++r) {
        const float iv = sigf(ga0[r]);
        const float fv = sigf(ga1[r]);
        const float gv = tanhf(ga2[r]);
        const float ov = sigf(ga3[r]);
        c[r] = fv*c[r] + iv*gv;
        h[r] = ov*tanhf(c[r]);
        Ast[(r + 8*half)*ASTR + 160 + jc] = (_Float16)h[r];
      }
    }
    __syncthreads();

    // phase 5: x_h2 = h_new @ W_hist^T + b_hist  (waves 8..11)
    if (wave >= 8 && wave < 12) {
      const int n = (wave - 8)*16 + lane16;
      const _Float16* arow = &Ast[lane16*ASTR + 160 + half*8];
      v16h a0 = frag16(arow +  0, arow + 16);
      v16h a1 = frag16(arow + 32, arow + 48);
      v16h a2 = frag16(arow + 64, arow + 80);
      v16h a3 = frag16(arow + 96, arow + 112);
      v8f acc = {bcol,bcol,bcol,bcol,bcol,bcol,bcol,bcol};
      acc = WMMA16(a0, Bh0, acc);
      acc = WMMA16(a1, Bh1, acc);
      acc = WMMA16(a2, Bh2, acc);
      acc = WMMA16(a3, Bh3, acc);
      #pragma unroll
      for (int r = 0; r < 8; ++r) xhb[(r + 8*half)*XHSTR + n] = acc[r];
    }
    __syncthreads();

    // phase 6: owners write imputations + accumulate loss partials;
    //          wave 15 preloads next-step d / w_surv into the other buffer.
    if (wave >= 8) {
      float num = 0.f, den = 0.f;
      #pragma unroll
      for (int q = 0; q < 4; ++q) {
        const int e = q*256 + ol, r = e >> 6, col = e & 63;
        const float m = mv[q], x = xv[q];
        const float xh2 = xhb[r*XHSTR + col];
        const float xf  = xfb[r*XHSTR + col];
        out[((size_t)(b0 + r)*T_ + t)*D_ + col] = m*x + (1.f - m)*(xh2 + xf);
        const float err = x - xh2;
        num += err*err*m*wrow[buf + r];
        den += m;
      }
      #pragma unroll
      for (int off = 16; off >= 1; off >>= 1) {
        num += __shfl_xor(num, off, 32);
        den += __shfl_xor(den, off, 32);
      }
      if (lane == 0) { atomicAdd(&lossb[0], num); atomicAdd(&lossb[1], den); }
    }
    if (wave == 15 && lane < 16 && t + 1 < T_) {
      const int b = b0 + lane;
      const int nb2 = ((t + 1) & 1) * 16;
      drow[nb2 + lane] = deltas[(size_t)b*T_ + (t + 1)];
      const float ag = ages[(size_t)b*T_ + (t + 1)];
      const int   cl = clusters[(size_t)b*T_ + (t + 1)];
      wrow[nb2 + lane] = __expf(-surv[cl] * ag);
    }
    __syncthreads();

    if (tid == 0) {   // publish per-(wg,t) loss partials; re-zero for next step
      ws[((size_t)blockIdx.x*T_ + t)*2    ] = lossb[0];
      ws[((size_t)blockIdx.x*T_ + t)*2 + 1] = lossb[1];
      lossb[0] = 0.f; lossb[1] = 0.f;
    }
  }
}

// Cross-workgroup loss reduction: loss = mean_t( sum_wg num / (sum_wg den + 1e-5) )
__global__ void rits_loss(const float* __restrict__ ws, float* __restrict__ out_loss) {
  __shared__ float red[256];
  const int tid = threadIdx.x;
  float s = 0.f;
  for (int t = tid; t < T_; t += 256) {
    float num = 0.f, den = 0.f;
    for (int g = 0; g < NWG; ++g) {
      num += ws[((size_t)g*T_ + t)*2];
      den += ws[((size_t)g*T_ + t)*2 + 1];
    }
    s += num / (den + 1e-5f);
  }
  red[tid] = s;
  __syncthreads();
  for (int off = 128; off > 0; off >>= 1) {
    if (tid < off) red[tid] += red[tid + off];
    __syncthreads();
  }
  if (tid == 0) out_loss[0] = red[0] / (float)T_;
}

extern "C" void kernel_launch(void* const* d_in, const int* in_sizes, int n_in,
                              void* d_out, int out_size, void* d_ws, size_t ws_size,
                              hipStream_t stream) {
  (void)in_sizes; (void)n_in; (void)out_size; (void)ws_size;
  const float* data     = (const float*)d_in[0];
  const float* masks    = (const float*)d_in[1];
  const float* deltas   = (const float*)d_in[2];
  const float* ages     = (const float*)d_in[3];
  const int*   clusters = (const int*)  d_in[4];
  const float* surv     = (const float*)d_in[5];
  const float* W_ih     = (const float*)d_in[6];
  const float* W_hh     = (const float*)d_in[7];
  const float* b_ih     = (const float*)d_in[8];
  const float* b_hh     = (const float*)d_in[9];
  const float* W_hist   = (const float*)d_in[10];
  const float* b_hist   = (const float*)d_in[11];
  const float* W_feat   = (const float*)d_in[12];
  const float* b_feat   = (const float*)d_in[13];
  const float* W_decay  = (const float*)d_in[14];
  const float* b_decay  = (const float*)d_in[15];
  float* out = (float*)d_out;
  float* ws  = (float*)d_ws;

  rits_main<<<dim3(NWG), dim3(512), (size_t)SMEM_BYTES, stream>>>(
      data, masks, deltas, ages, clusters, surv,
      W_ih, W_hh, b_ih, b_hh, W_hist, b_hist, W_feat, b_feat,
      W_decay, b_decay, out, ws);
  rits_loss<<<dim3(1), dim3(256), 0, stream>>>(ws, out + (size_t)B_*T_*D_);
}
